// FakeMoE_19619410608456
// MI455X (gfx1250) — matrix operand order repaired and checked by reading, not compiled, not run on hardware
//
#include <hip/hip_runtime.h>

// FakeMoE: T=8192 tokens, H=4096, E=8 experts, top-2 routing (combine weight 1.0).
// Pipeline:
//   memset(out, counts) ->
//   gate_topk_cvt: logits + top-2 + per-expert token lists, AND writes bf16 copy
//                  of x into workspace (fused conversion, x is already streamed) ->
//   moe_expert_gemm: per-expert gathered GEMM, software-pipelined + LDS
//                  double-buffered; A tile copied via GLOBAL_LOAD_ASYNC_TO_LDS
//                  (bf16, no conversion needed), B = fp32 weights converted to
//                  bf16 while staging; bf16 WMMA math, atomicAdd f32 combine
//                  (exactly 2 adds/element -> deterministic).

#define TOKENS 8192
#define HDIM   4096
#define NEXP   8

#define TM 128           // tokens per block tile
#define TN 128           // output features per block tile
#define KC 64            // K per stage == two bf16 WMMA K-steps
#define LDK 72           // padded LDS row stride in bf16 elems (144B, 16B-aligned, conflict-free)

typedef __attribute__((ext_vector_type(16))) __bf16 bf16x16;
typedef __attribute__((ext_vector_type(8)))  __bf16 bf16x8;
typedef __attribute__((ext_vector_type(2)))  __bf16 bf16x2;
typedef __attribute__((ext_vector_type(8)))  float  f32x8;
typedef __attribute__((ext_vector_type(4)))  int    v4i;

#if __has_builtin(__builtin_amdgcn_global_load_async_to_lds_b128) && \
    __has_builtin(__builtin_amdgcn_s_wait_asynccnt)
#define USE_ASYNC_A 1
#else
#define USE_ASYNC_A 0
#endif

__device__ __forceinline__ unsigned int pack2_bf16(float a, float b) {
#if __has_builtin(__builtin_amdgcn_cvt_pk_bf16_f32)
    bf16x2 r = __builtin_amdgcn_cvt_pk_bf16_f32(a, b);
    return __builtin_bit_cast(unsigned int, r);
#else
    // round-to-nearest-even fp32 -> bf16, packed pair
    unsigned int ua = __float_as_uint(a);
    unsigned int ub = __float_as_uint(b);
    ua += 0x7FFFu + ((ua >> 16) & 1u);
    ub += 0x7FFFu + ((ub >> 16) & 1u);
    return (ua >> 16) | (ub & 0xFFFF0000u);
#endif
}

// ---------------------------------------------------------------------------
// Kernel 1: gating + top-2 + token lists, fused with x -> bf16 conversion.
// ---------------------------------------------------------------------------
__global__ __launch_bounds__(256) void gate_topk_cvt_kernel(
    const float* __restrict__ x,
    const float* __restrict__ gate_w,
    int* __restrict__ counts,
    int* __restrict__ lists,
    unsigned int* __restrict__ xb) {     // [T, H/2] packed bf16 pairs
    const int wid   = threadIdx.x >> 5;
    const int lane  = threadIdx.x & 31;
    const int token = blockIdx.x * 8 + wid;
    if (token >= TOKENS) return;

    float acc[NEXP];
#pragma unroll
    for (int e = 0; e < NEXP; ++e) acc[e] = 0.0f;

    const float2* xr2 = (const float2*)(x + (size_t)token * HDIM);
    unsigned int* xbr = xb + (size_t)token * (HDIM / 2);
    for (int i = lane; i < HDIM / 2; i += 32) {
        const float2 xv = xr2[i];
        xbr[i] = pack2_bf16(xv.x, xv.y);
#pragma unroll
        for (int e = 0; e < NEXP; ++e) {
            const float2 gw = ((const float2*)(gate_w + e * HDIM))[i];
            acc[e] = fmaf(xv.x, gw.x, fmaf(xv.y, gw.y, acc[e]));
        }
    }
#pragma unroll
    for (int e = 0; e < NEXP; ++e) {
#pragma unroll
        for (int off = 16; off > 0; off >>= 1)
            acc[e] += __shfl_xor(acc[e], off, 32);
    }

    if (lane == 0) {
        int b1 = 0; float v1 = acc[0];
#pragma unroll
        for (int e = 1; e < NEXP; ++e)
            if (acc[e] > v1) { v1 = acc[e]; b1 = e; }
        int b2 = -1; float v2 = -3.4e38f;
#pragma unroll
        for (int e = 0; e < NEXP; ++e)
            if (e != b1 && acc[e] > v2) { v2 = acc[e]; b2 = e; }

        int s1 = atomicAdd(&counts[b1], 1);
        lists[b1 * TOKENS + s1] = token;
        int s2 = atomicAdd(&counts[b2], 1);
        lists[b2 * TOKENS + s2] = token;
    }
}

// ---------------------------------------------------------------------------
// Kernel 2: per-expert gathered GEMM, software-pipelined, double-buffered LDS.
// Grid: (HDIM/TN, TOKENS/TM, NEXP); expert slowest so W[e] (64MB) stays in L2.
// ---------------------------------------------------------------------------
__global__ __launch_bounds__(256) void moe_expert_gemm_kernel(
    const unsigned int* __restrict__ xb,     // [T, H/2] packed bf16 pairs
    const float* __restrict__ expert_w,
    const int* __restrict__ counts,
    const int* __restrict__ lists,
    float* __restrict__ out) {
    const int e   = blockIdx.z;
    const int n_e = counts[e];
    const int row_base = blockIdx.y * TM;
    if (row_base >= n_e) return;           // inactive row tile for this expert
    const int col_base = blockIdx.x * TN;

    __shared__ unsigned short As[2][TM][LDK];  // [buf][token-row][k] bf16
    __shared__ unsigned short Bs[2][TN][LDK];  // [buf][out-col ][k] bf16

    const int tid    = threadIdx.x;
    const int lane   = tid & 31;
    const int wid    = tid >> 5;
    const int wave_m = wid >> 2;            // 0..1  -> 64 rows each
    const int wave_n = wid & 3;             // 0..3  -> 32 cols each
    const int half   = lane >> 4;           // 0/1 (lane group)
    const int l16    = lane & 15;

    // Loader assignment: thread owns (row = tid/2, K-half = tid&1) of both tiles.
    // Per stage each thread moves 32 bf16 (64B) of A and 32 fp32 -> 32 bf16 of B.
    const int ld_row  = tid >> 1;           // 0..127
    const int ld_half = tid & 1;            // 0/1 -> K element offset 0 or 32

    const int* mylist = lists + e * TOKENS;
    const bool a_valid = (row_base + ld_row) < n_e;
    const int  a_tok   = a_valid ? mylist[row_base + ld_row] : 0;
    const unsigned int* a_src = xb + (size_t)a_tok * (HDIM / 2) + ld_half * 16;
    const float* b_src = expert_w +
        ((size_t)e * HDIM + (size_t)(col_base + ld_row)) * HDIM + ld_half * 32;

    f32x8 c[4][2];
#pragma unroll
    for (int fm = 0; fm < 4; ++fm)
#pragma unroll
        for (int fn = 0; fn < 2; ++fn)
#pragma unroll
            for (int v = 0; v < 8; ++v) c[fm][fn][v] = 0.0f;

    // stage registers (prefetched global data)
    uint4 av[4];
    unsigned int bpk[16];

    auto load_A = [&](int kc) {
        if (a_valid) {
            const uint4* ap = (const uint4*)(a_src + kc / 2);
            av[0] = ap[0]; av[1] = ap[1]; av[2] = ap[2]; av[3] = ap[3];
        } else {
            const uint4 z = make_uint4(0u, 0u, 0u, 0u);
            av[0] = z; av[1] = z; av[2] = z; av[3] = z;
        }
    };
    auto store_A = [&](int b) {
        uint4* ad = (uint4*)&As[b][ld_row][ld_half * 32];
        ad[0] = av[0]; ad[1] = av[1]; ad[2] = av[2]; ad[3] = av[3];
    };
    auto load_B = [&](int kc) {
        const float4* q = (const float4*)(b_src + kc);
#pragma unroll
        for (int j = 0; j < 8; ++j) {
            const float4 g = q[j];
            bpk[2 * j]     = pack2_bf16(g.x, g.y);
            bpk[2 * j + 1] = pack2_bf16(g.z, g.w);
        }
    };
    auto store_B = [&](int b) {
        uint4* bd = (uint4*)&Bs[b][ld_row][ld_half * 32];
        bd[0] = make_uint4(bpk[0],  bpk[1],  bpk[2],  bpk[3]);
        bd[1] = make_uint4(bpk[4],  bpk[5],  bpk[6],  bpk[7]);
        bd[2] = make_uint4(bpk[8],  bpk[9],  bpk[10], bpk[11]);
        bd[3] = make_uint4(bpk[12], bpk[13], bpk[14], bpk[15]);
    };

    // prologue: stage 0 into buffer 0 (register path)
    load_A(0);
    load_B(0);
    store_A(0);
    store_B(0);
#if USE_ASYNC_A
    if (!a_valid) {
        // tail rows are never rewritten by the async path: zero them in buf 1 too
        const uint4 z = make_uint4(0u, 0u, 0u, 0u);
        uint4* ad = (uint4*)&As[1][ld_row][ld_half * 32];
        ad[0] = z; ad[1] = z; ad[2] = z; ad[3] = z;
    }
#endif

    int buf = 0;
    for (int kc = 0; kc < HDIM; kc += KC) {
        __syncthreads();     // publishes LDS[buf]; guarantees LDS[buf^1] reads of
                             // the previous stage are complete (safe to overwrite)
        const bool has_next = (kc + KC) < HDIM;
        if (has_next) {
#if USE_ASYNC_A
            if (a_valid) {
                // direct global->LDS copy of next A slice (bf16, 64B/thread),
                // tracked on ASYNCcnt; imm offset applies to both addresses.
                v4i* gp = (v4i*)(a_src + (kc + KC) / 2);
                v4i* lp = (v4i*)&As[buf ^ 1][ld_row][ld_half * 32];
                __builtin_amdgcn_global_load_async_to_lds_b128(gp, lp, 0,  0);
                __builtin_amdgcn_global_load_async_to_lds_b128(gp, lp, 16, 0);
                __builtin_amdgcn_global_load_async_to_lds_b128(gp, lp, 32, 0);
                __builtin_amdgcn_global_load_async_to_lds_b128(gp, lp, 48, 0);
            }
#else
            load_A(kc + KC);
#endif
            load_B(kc + KC);              // overlaps with WMMAs below
            if (kc + 2 * KC < HDIM)
                __builtin_prefetch(b_src + kc + 2 * KC, 0, 3);  // global_prefetch_b8
        }

        // ---- two WMMA K-steps on LDS[buf] ----
#pragma unroll
        for (int ks = 0; ks < KC; ks += 32) {
            bf16x16 bfrag[2];
#pragma unroll
            for (int fn = 0; fn < 2; ++fn) {
                const int n = wave_n * 32 + fn * 16 + l16;
                union { bf16x16 v; bf16x8 h[2]; } u;
                // B 32x16 bf16: lanes 0-15 K={0..7,16..23}, lanes 16-31 K={8..15,24..31}
                u.h[0] = *(const bf16x8*)&Bs[buf][n][ks + half * 8];
                u.h[1] = *(const bf16x8*)&Bs[buf][n][ks + 16 + half * 8];
                bfrag[fn] = u.v;
            }
#pragma unroll
            for (int fm = 0; fm < 4; ++fm) {
                const int m = wave_m * 64 + fm * 16 + l16;
                union { bf16x16 v; bf16x8 h[2]; } u;
                // A 16x32 bf16: lanes 0-15 K={0..7,16..23}, lanes 16-31 K={8..15,24..31}
                u.h[0] = *(const bf16x8*)&As[buf][m][ks + half * 8];
                u.h[1] = *(const bf16x8*)&As[buf][m][ks + 16 + half * 8];
                const bf16x16 afrag = u.v;
#pragma unroll
                for (int fn = 0; fn < 2; ++fn) {
                    c[fm][fn] = __builtin_amdgcn_wmma_f32_16x16x32_bf16(
                        false, afrag, false, bfrag[fn],
                        (short)0, c[fm][fn], false, false);
                }
            }
        }

        if (has_next) {
#if !USE_ASYNC_A
            store_A(buf ^ 1);
#endif
            store_B(buf ^ 1);
#if USE_ASYNC_A
            __builtin_amdgcn_s_wait_asynccnt(0);   // A slice landed in LDS
#endif
        }
        buf ^= 1;
    }

    // ---- epilogue: combine via atomic f32 add (exactly 2 adds per element) ----
#pragma unroll
    for (int fm = 0; fm < 4; ++fm) {
        const int m_tile = wave_m * 64 + fm * 16;
#pragma unroll
        for (int fn = 0; fn < 2; ++fn) {
            const int col = col_base + wave_n * 32 + fn * 16 + l16;
#pragma unroll
            for (int v = 0; v < 8; ++v) {
                // C/D layout: VGPR v -> M = v + 8*half, N = lane&15
                const int gm = row_base + m_tile + v + 8 * half;
                if (gm < n_e) {
                    const int tok = mylist[gm];
                    atomicAdd(&out[(size_t)tok * HDIM + col], c[fm][fn][v]);
                }
            }
        }
    }
}

// ---------------------------------------------------------------------------
extern "C" void kernel_launch(void* const* d_in, const int* in_sizes, int n_in,
                              void* d_out, int out_size, void* d_ws, size_t ws_size,
                              hipStream_t stream) {
    (void)in_sizes; (void)n_in; (void)ws_size;
    const float* x        = (const float*)d_in[0];   // [T, H]
    const float* gate_w   = (const float*)d_in[1];   // [E, H]
    const float* expert_w = (const float*)d_in[2];   // [E, H, H]
    float*       out      = (float*)d_out;           // [T, H]

    // workspace layout: [counts: 8 ints][pad][lists: E*T ints][pad][xb: T*H/2 uints]
    int*          counts = (int*)d_ws;
    int*          lists  = (int*)((char*)d_ws + 256);
    unsigned int* xb     = (unsigned int*)((char*)d_ws + (512u * 1024u));

    (void)hipMemsetAsync(out, 0, (size_t)out_size * sizeof(float), stream);
    (void)hipMemsetAsync(counts, 0, NEXP * sizeof(int), stream);

    gate_topk_cvt_kernel<<<TOKENS / 8, 256, 0, stream>>>(x, gate_w, counts, lists, xb);

    dim3 grid(HDIM / TN, TOKENS / TM, NEXP);
    moe_expert_gemm_kernel<<<grid, 256, 0, stream>>>(xb, expert_w, counts, lists, out);
}
